// PSFNet_13168369729643
// MI455X (gfx1250) — matrix operand order repaired and checked by reading, not compiled, not run on hardware
//
#include <hip/hip_runtime.h>

// ---------------- problem constants ----------------
constexpr int B_      = 16;
constexpr int N_VEC   = 4096;
constexpr int D_      = 256;
constexpr int C_      = 256;
constexpr int N_W     = 12;
constexpr int N_LINKS = 13;
constexpr int N_CLASS = 10;
constexpr int ROWS    = B_ * N_VEC;          // 65536

// ---------------- WMMA types ----------------
typedef __bf16  v16bf __attribute__((ext_vector_type(16)));
typedef float   v8f   __attribute__((ext_vector_type(8)));
typedef unsigned int u32x4 __attribute__((ext_vector_type(4)));

union FragBF {
    u32x4 u[2];
    v16bf v;
};

// ---------------- CDNA5 async-to-LDS helpers (ISA 15.18.3 / ASYNCcnt) ----------------
// VDST VGPR carries the LDS byte address; generic shared pointers keep the LDS
// offset in their low 32 bits (ISA 10.2 LDS aperture).
__device__ __forceinline__ void async_load_b128(unsigned lds_byte_addr, const void* gaddr) {
    asm volatile("global_load_async_to_lds_b128 %0, %1, off"
                 :: "v"(lds_byte_addr), "v"((unsigned long long)(size_t)gaddr)
                 : "memory");
}
__device__ __forceinline__ void wait_asynccnt0() {
    asm volatile("s_wait_asynccnt 0" ::: "memory");
}

// A-fragment (16x32 bf16, M rows striped over lanes) from a row-major matrix
// (leading dim `ld` elements). lane<16 holds K=k0..k0+15 (contiguous);
// lane>=16 holds K=k0+16..k0+31.
__device__ __forceinline__ v16bf load_frag_a(const __bf16* M, int ld, int m0, int k0, int lane) {
    int row = m0 + (lane & 15);
    int kA  = k0 + ((lane >> 4) << 4);          // +16 for upper half-wave
    const __bf16* p = M + (size_t)row * ld + kA;
    FragBF f;
    f.u[0] = *(const u32x4*)(p);                // K = kA .. kA+7
    f.u[1] = *(const u32x4*)(p + 8);            // K = kA+8 .. kA+15
    return f.v;
}

// B-fragment (32x16 bf16, N cols striped over lanes) from a *transposed*
// (col-major, i.e. row-major NxK) weight matrix: lane n holds col n0+n,
// lanes 0-15 K=k0..k0+15, lanes 16-31 K=k0+16..k0+31 (contiguous 32 B).
__device__ __forceinline__ v16bf load_frag_b(const __bf16* WT, int ld, int n0, int k0, int lane) {
    int col = n0 + (lane & 15);
    int kB  = k0 + ((lane >> 4) << 4);          // +16 for upper half-wave
    const __bf16* p = WT + (size_t)col * ld + kB;
    FragBF f;
    f.u[0] = *(const u32x4*)(p);
    f.u[1] = *(const u32x4*)(p + 8);
    return f.v;
}

__device__ __forceinline__ float gelu_exact(float x) {
    return 0.5f * x * (1.0f + erff(x * 0.70710678118654752f));
}

// ---------------- kernel 1: pack / transpose weights to bf16 ----------------
// dest-indexed segments:
//   w1T  : [C][D]        <- g_w1[k*C + c]
//   w2T  : [C][C]        <- g_w2[k*C + c]
//   f1T  : [N_W][D][D]   <- fs_w1[m][k*D + c]
//   f2T  : [N_W][16][D]  <- fs_w2[m][k*N_LINKS + l]  (l>=13 zero-padded)
__global__ __launch_bounds__(256) void pack_weights_kernel(
    const float* __restrict__ gw1, const float* __restrict__ gw2,
    const float* __restrict__ fw1, const float* __restrict__ fw2,
    __bf16* __restrict__ w1T, __bf16* __restrict__ w2T,
    __bf16* __restrict__ f1T, __bf16* __restrict__ f2T)
{
    int i = blockIdx.x * 256 + threadIdx.x;
    const int S0 = D_ * C_;                   // 65536
    const int S1 = S0 + C_ * C_;              // 131072
    const int S2 = S1 + N_W * D_ * D_;        // 917504
    const int S3 = S2 + N_W * 16 * D_;        // 966656
    if (i < S0) {
        int c = i >> 8, k = i & 255;
        w1T[i] = (__bf16)gw1[k * C_ + c];
    } else if (i < S1) {
        int j = i - S0; int c = j >> 8, k = j & 255;
        w2T[j] = (__bf16)gw2[k * C_ + c];
    } else if (i < S2) {
        int j = i - S1; int m = j >> 16; int r = j & 65535;
        int c = r >> 8, k = r & 255;
        f1T[j] = (__bf16)fw1[(size_t)m * D_ * D_ + k * D_ + c];
    } else if (i < S3) {
        int j = i - S2; int m = j >> 12; int r = j & 4095;
        int l = r >> 8, k = r & 255;
        f2T[j] = (l < N_LINKS) ? (__bf16)fw2[(size_t)m * D_ * N_LINKS + k * N_LINKS + l]
                               : (__bf16)0.0f;
    }
}

// ---------------- kernel 2: embedding -> bf16 activations ----------------
__global__ __launch_bounds__(256) void embed_kernel(
    const int* __restrict__ tokens, const float* __restrict__ emb,
    const float* __restrict__ pos, __bf16* __restrict__ out)
{
    int i = blockIdx.x * 256 + threadIdx.x;           // < ROWS*256
    int c = i & 255;
    int row = i >> 8;
    int n = row & (N_VEC - 1);
    int tok = tokens[row];
    out[i] = (__bf16)(emb[tok * D_ + c] + pos[n * D_ + c]);
}

// stage a contiguous 32x256 bf16 tile (16 KB) into LDS via async DMA:
// 4 rounds x 256 threads x 16 B, fully coalesced, tracked on ASYNCcnt.
__device__ __forceinline__ void stage_x_tile(__bf16* Xs, const __bf16* Xbf,
                                             int rowBase, int tid)
{
    const char* src = (const char*)(Xbf + (size_t)rowBase * D_);
    unsigned dstBase = (unsigned)(size_t)(void*)Xs;
    #pragma unroll
    for (int r = 0; r < 4; ++r) {
        int byteOff = (r * 256 + tid) * 16;
        async_load_b128(dstBase + byteOff, src + byteOff);
    }
    wait_asynccnt0();
    __syncthreads();
}

// ---------------- kernel 3: fused V = gelu(X@W1+b1)@W2 + b2 ----------------
// Block: 32 rows, 256 threads = 8 waves as 2(M) x 4(N).
__global__ __launch_bounds__(256) void gemm_fused_g_kernel(
    const __bf16* __restrict__ Xbf,   // [ROWS][D]
    const __bf16* __restrict__ w1T,   // [C][D]
    const float*  __restrict__ b1,    // [C]
    const __bf16* __restrict__ w2T,   // [C][C]
    const float*  __restrict__ b2,    // [C]
    float* __restrict__ V)            // [ROWS][C]
{
    __shared__ __bf16 Xs[32 * 256];
    __shared__ __bf16 Hs[32 * 256];
    const int tid  = threadIdx.x;
    const int lane = tid & 31;
    const int wid  = tid >> 5;
    const int mi   = wid & 1;       // 16-row sub-block
    const int ni   = wid >> 1;      // 64-col group
    const int rowBase = blockIdx.x * 32;

    stage_x_tile(Xs, Xbf, rowBase, tid);

    // ---- stage 1: H = gelu(X @ W1 + b1) ----
    v8f acc[4] = { {}, {}, {}, {} };
    for (int k0 = 0; k0 < D_; k0 += 32) {
        v16bf a = load_frag_a(&Xs[0], D_, mi * 16, k0, lane);
        #pragma unroll
        for (int t = 0; t < 4; ++t) {
            v16bf b = load_frag_b(w1T, D_, ni * 64 + t * 16, k0, lane);
            acc[t] = __builtin_amdgcn_wmma_f32_16x16x32_bf16(
                         false, a, false, b, (short)0, acc[t], false, false);
        }
    }
    {   // epilogue: bias + gelu, bf16 into LDS (row-major, per C/D layout)
        const int nCol = lane & 15;
        const int mOff = (lane >> 4) << 3;           // 0 or 8
        #pragma unroll
        for (int t = 0; t < 4; ++t) {
            int col = ni * 64 + t * 16 + nCol;
            float bias = b1[col];
            #pragma unroll
            for (int r = 0; r < 8; ++r) {
                float x = acc[t][r] + bias;
                Hs[(mi * 16 + mOff + r) * 256 + col] = (__bf16)gelu_exact(x);
            }
        }
    }
    __syncthreads();

    // ---- stage 2: V = H @ W2 + b2 ----
    v8f acc2[4] = { {}, {}, {}, {} };
    for (int k0 = 0; k0 < C_; k0 += 32) {
        v16bf a = load_frag_a(&Hs[0], 256, mi * 16, k0, lane);
        #pragma unroll
        for (int t = 0; t < 4; ++t) {
            v16bf b = load_frag_b(w2T, C_, ni * 64 + t * 16, k0, lane);
            acc2[t] = __builtin_amdgcn_wmma_f32_16x16x32_bf16(
                          false, a, false, b, (short)0, acc2[t], false, false);
        }
    }
    {
        const int nCol = lane & 15;
        const int mOff = (lane >> 4) << 3;
        #pragma unroll
        for (int t = 0; t < 4; ++t) {
            int col = ni * 64 + t * 16 + nCol;
            float bias = b2[col];
            #pragma unroll
            for (int r = 0; r < 8; ++r) {
                size_t row = (size_t)(rowBase + mi * 16 + mOff + r);
                V[row * C_ + col] = acc2[t][r] + bias;
            }
        }
    }
}

// ---------------- kernel 4: W_all[m] = gelu(X@fs_w1[m]+b)@fs_w2[m] + b ----------------
// grid = (ROWS/32, N_W). Stage 1 identical; stage 2 is a single 16-col tile.
__global__ __launch_bounds__(256) void gemm_fused_fs_kernel(
    const __bf16* __restrict__ Xbf,     // [ROWS][D]
    const __bf16* __restrict__ f1T,     // [N_W][D][D]
    const float*  __restrict__ fb1,     // [N_W][D]
    const __bf16* __restrict__ f2T,     // [N_W][16][D]
    const float*  __restrict__ fb2,     // [N_W][N_LINKS]
    float* __restrict__ Wall)           // [N_W][ROWS][16]
{
    __shared__ __bf16 Xs[32 * 256];
    __shared__ __bf16 Hs[32 * 256];
    const int tid  = threadIdx.x;
    const int lane = tid & 31;
    const int wid  = tid >> 5;
    const int mi   = wid & 1;
    const int ni   = wid >> 1;
    const int rowBase = blockIdx.x * 32;
    const int m = blockIdx.y;

    const __bf16* w1T = f1T + (size_t)m * D_ * D_;
    const float*  b1  = fb1 + (size_t)m * D_;

    stage_x_tile(Xs, Xbf, rowBase, tid);

    v8f acc[4] = { {}, {}, {}, {} };
    for (int k0 = 0; k0 < D_; k0 += 32) {
        v16bf a = load_frag_a(&Xs[0], D_, mi * 16, k0, lane);
        #pragma unroll
        for (int t = 0; t < 4; ++t) {
            v16bf b = load_frag_b(w1T, D_, ni * 64 + t * 16, k0, lane);
            acc[t] = __builtin_amdgcn_wmma_f32_16x16x32_bf16(
                         false, a, false, b, (short)0, acc[t], false, false);
        }
    }
    {
        const int nCol = lane & 15;
        const int mOff = (lane >> 4) << 3;
        #pragma unroll
        for (int t = 0; t < 4; ++t) {
            int col = ni * 64 + t * 16 + nCol;
            float bias = b1[col];
            #pragma unroll
            for (int r = 0; r < 8; ++r) {
                float x = acc[t][r] + bias;
                Hs[(mi * 16 + mOff + r) * 256 + col] = (__bf16)gelu_exact(x);
            }
        }
    }
    __syncthreads();

    if (wid < 2) {   // stage 2: 2 waves, each one 16x16 output tile (13 live cols)
        const __bf16* w2T = f2T + (size_t)m * 16 * D_;
        v8f acc2 = {};
        for (int k0 = 0; k0 < D_; k0 += 32) {
            v16bf a = load_frag_a(&Hs[0], 256, wid * 16, k0, lane);
            v16bf b = load_frag_b(w2T, D_, 0, k0, lane);
            acc2 = __builtin_amdgcn_wmma_f32_16x16x32_bf16(
                       false, a, false, b, (short)0, acc2, false, false);
        }
        const int nCol = lane & 15;
        const int mOff = (lane >> 4) << 3;
        float bias = (nCol < N_LINKS) ? fb2[m * N_LINKS + nCol] : 0.0f;
        #pragma unroll
        for (int r = 0; r < 8; ++r) {
            size_t row = (size_t)(rowBase + wid * 16 + mOff + r);
            Wall[((size_t)m * ROWS + row) * 16 + nCol] = acc2[r] + bias;
        }
    }
}

// ---------------- kernel 5: mix step V' = sum_l W[l]*V[cols[l]] + res ----------------
// one block per (b,n) row; thread = output channel (C=256).
__global__ __launch_bounds__(256) void mix_kernel(
    const float* __restrict__ Vcur, float* __restrict__ Vnxt,
    const float* __restrict__ res,  const float* __restrict__ Wm,  // [ROWS][16]
    const int*   __restrict__ cols)
{
    __shared__ float ws[N_LINKS];
    __shared__ int   rbase[N_LINKS];
    const int row = blockIdx.x;               // b*N_VEC + n
    const int n   = row & (N_VEC - 1);
    const int bo  = row & ~(N_VEC - 1);       // b*N_VEC
    const int t   = threadIdx.x;
    if (t < N_LINKS) {
        ws[t]    = Wm[(size_t)row * 16 + t];
        rbase[t] = bo + cols[n * N_LINKS + t];
    }
    __syncthreads();
    float acc = res[(size_t)row * C_ + t];
    #pragma unroll
    for (int l = 0; l < N_LINKS; ++l)
        acc += ws[l] * Vcur[(size_t)rbase[l] * C_ + t];
    Vnxt[(size_t)row * C_ + t] = acc;
}

// ---------------- kernel 6: out = cls @ fin_w + fin_b ----------------
__global__ __launch_bounds__(256) void final_kernel(
    const float* __restrict__ Vf, const float* __restrict__ fw,
    const float* __restrict__ fb, float* __restrict__ out)
{
    int t = threadIdx.x;
    if (t >= B_ * N_CLASS) return;
    int b = t / N_CLASS, j = t % N_CLASS;
    const float* v = Vf + (size_t)(b * N_VEC) * C_;   // row (b, 0)
    float acc = fb[j];
    for (int c = 0; c < C_; ++c) acc += v[c] * fw[c * N_CLASS + j];
    out[t] = acc;
}

// ---------------- host-side orchestration ----------------
extern "C" void kernel_launch(void* const* d_in, const int* in_sizes, int n_in,
                              void* d_out, int out_size, void* d_ws, size_t ws_size,
                              hipStream_t stream)
{
    const int*   tokens = (const int*)  d_in[0];
    const int*   cols   = (const int*)  d_in[1];
    const float* emb    = (const float*)d_in[2];
    const float* pos    = (const float*)d_in[3];
    const float* g_w1   = (const float*)d_in[4];
    const float* g_b1   = (const float*)d_in[5];
    const float* g_w2   = (const float*)d_in[6];
    const float* g_b2   = (const float*)d_in[7];
    const float* fs_w1  = (const float*)d_in[8];
    const float* fs_b1  = (const float*)d_in[9];
    const float* fs_w2  = (const float*)d_in[10];
    const float* fs_b2  = (const float*)d_in[11];
    const float* fin_w  = (const float*)d_in[12];
    const float* fin_b  = (const float*)d_in[13];
    float* out = (float*)d_out;
    char*  ws  = (char*)d_ws;

    // workspace carve-up (bytes)
    size_t off = 0;
    __bf16* dataBf = (__bf16*)(ws + off); off += (size_t)ROWS * D_ * 2;        // 32 MB
    __bf16* w1T    = (__bf16*)(ws + off); off += (size_t)C_ * D_ * 2;
    __bf16* w2T    = (__bf16*)(ws + off); off += (size_t)C_ * C_ * 2;
    __bf16* f1T    = (__bf16*)(ws + off); off += (size_t)N_W * D_ * D_ * 2;
    __bf16* f2T    = (__bf16*)(ws + off); off += (size_t)N_W * 16 * D_ * 2;
    float*  V0     = (float*) (ws + off); off += (size_t)ROWS * C_ * 4;        // res
    float*  V1     = (float*) (ws + off); off += (size_t)ROWS * C_ * 4;
    float*  V2     = (float*) (ws + off); off += (size_t)ROWS * C_ * 4;
    float*  Wall   = (float*) (ws + off); off += (size_t)N_W * ROWS * 16 * 4;

    // 1) weight packing (966656 dest elements)
    pack_weights_kernel<<<(966656 + 255) / 256, 256, 0, stream>>>(
        g_w1, g_w2, fs_w1, fs_w2, w1T, w2T, f1T, f2T);

    // 2) embedding -> bf16 activations
    embed_kernel<<<(ROWS * D_) / 256, 256, 0, stream>>>(tokens, emb, pos, dataBf);

    // 3) V0 = gelu(X@W1+b1)@W2 + b2   (res == V0)
    gemm_fused_g_kernel<<<ROWS / 32, 256, 0, stream>>>(
        dataBf, w1T, g_b1, w2T, g_b2, V0);

    // 4) all link-weight matrices W_all[m] in one batched pass
    gemm_fused_fs_kernel<<<dim3(ROWS / 32, N_W), 256, 0, stream>>>(
        dataBf, f1T, fs_b1, f2T, fs_b2, Wall);

    // 5) 12 sequential mix steps, ping-pong between V1/V2 (V0 stays as res)
    const float* cur = V0;
    float* nxt = V1;
    for (int m = 0; m < N_W; ++m) {
        mix_kernel<<<ROWS, 256, 0, stream>>>(
            cur, nxt, V0, Wall + (size_t)m * ROWS * 16, cols);
        cur = nxt;
        nxt = (cur == V1) ? V2 : V1;
    }

    // 6) classifier head
    final_kernel<<<1, 256, 0, stream>>>(cur, fin_w, fin_b, out);
}
